// QuantBasicTransformerBlock_84567906058495
// MI455X (gfx1250) — compile-verified, matrix operand order checked by reading, not run on hardware
//
#include <hip/hip_runtime.h>
#include <math.h>

// ---------- types ----------
typedef __bf16 bf16_t;
typedef __attribute__((ext_vector_type(16))) __bf16 v16bf;
typedef __attribute__((ext_vector_type(8)))  __bf16 v8bf;
typedef __attribute__((ext_vector_type(8)))  float  v8f;

// ---------- problem constants ----------
#define B_   8
#define N_   1024
#define C_   1280
#define H_   20
#define DH_  64
#define M_   77
#define MP_  96        // padded key count for cross attention (3 tiles of 32)
#define CC_  768
#define FF_  5120
#define FF2_ 10240
#define R_   8192      // B_*N_
#define RC_  616       // B_*M_

// ---------- WMMA helpers ----------
__device__ __forceinline__ v8f wmma_bf(v16bf a, v16bf b, v8f c) {
  // D(f32,16x16) = A(bf16,16x32) * B(bf16,32x16) + C
  return __builtin_amdgcn_wmma_f32_16x16x32_bf16(false, a, false, b, (short)0, c, false, false);
}

__device__ __forceinline__ v16bf ldfrag(const bf16_t* p) {
  // per-lane 16 bf16 = two contiguous 16B chunks at K offsets {+0, +16}
  union { v16bf v; v8bf h[2]; } u;
  u.h[0] = *reinterpret_cast<const v8bf*>(p);
  u.h[1] = *reinterpret_cast<const v8bf*>(p + 16);
  return u.v;
}

__device__ __forceinline__ float gelu_tanh(float x) {
  float x3 = x * x * x;
  return 0.5f * x * (1.0f + tanhf(0.7978845608028654f * (x + 0.044715f * x3)));
}

// ---------- init scalars ----------
__global__ void init_scal_kernel(float* s) {
  if (threadIdx.x < 16) s[threadIdx.x] = 0.0f;
}

// ---------- weight transpose + f32->bf16 : WT[n][k] = W[k][n] ----------
__global__ __launch_bounds__(256) void wcvt_kernel(const float* __restrict__ W,
                                                   bf16_t* __restrict__ WT,
                                                   int K, int Nc) {
  size_t idx = (size_t)blockIdx.x * 256 + threadIdx.x;
  size_t total = (size_t)K * Nc;
  if (idx >= total) return;
  int k = (int)(idx / Nc);
  int n = (int)(idx % Nc);
  WT[(size_t)n * K + k] = (bf16_t)W[idx];
}

// ---------- elementwise f32 -> bf16 ----------
__global__ __launch_bounds__(256) void cvt_kernel(const float* __restrict__ in,
                                                  bf16_t* __restrict__ out, size_t n) {
  size_t idx = (size_t)blockIdx.x * 256 + threadIdx.x;
  if (idx < n) out[idx] = (bf16_t)in[idx];
}

// ---------- fq_sym quantize (per-tensor symmetric) f32 -> bf16 ----------
__global__ __launch_bounds__(256) void quant_cvt_kernel(const float* __restrict__ in,
                                                        bf16_t* __restrict__ out,
                                                        const float* __restrict__ amax,
                                                        size_t n) {
  size_t idx = (size_t)blockIdx.x * 256 + threadIdx.x;
  if (idx >= n) return;
  float delta = fmaxf(amax[0], 1e-8f) / 127.0f;
  float q = rintf(in[idx] / delta);
  q = fminf(fmaxf(q, -128.0f), 127.0f) * delta;
  out[idx] = (bf16_t)q;
}

// ---------- layernorm (row-wise, C cols) f32 -> bf16 ----------
__global__ __launch_bounds__(256) void ln_kernel(const float* __restrict__ x,
                                                 const float* __restrict__ g,
                                                 const float* __restrict__ b,
                                                 bf16_t* __restrict__ out, int C) {
  __shared__ float rs[256], rs2[256];
  int tid = threadIdx.x;
  size_t base = (size_t)blockIdx.x * C;
  float s = 0.f, s2 = 0.f;
  for (int c = tid; c < C; c += 256) { float v = x[base + c]; s += v; s2 += v * v; }
  rs[tid] = s; rs2[tid] = s2;
  for (int o = 128; o > 0; o >>= 1) {
    __syncthreads();
    if (tid < o) { rs[tid] += rs[tid + o]; rs2[tid] += rs2[tid + o]; }
  }
  __syncthreads();
  float mu = rs[0] / C;
  float var = rs2[0] / C - mu * mu;
  float inv = rsqrtf(var + 1e-5f);
  for (int c = tid; c < C; c += 256)
    out[base + c] = (bf16_t)((x[base + c] - mu) * inv * g[c] + b[c]);
}

// ---------- WMMA GEMM: C[f32, MxN] = A[bf16, MxK] * BT[bf16, NxK]^T (+bias +resid) ----------
// Block tile 256x128 (8 waves, 4x2), wave tile 64x64 (4x4 WMMA accumulators).
// 8 fragment loads (16x b128) per 16 WMMAs -> 1 load/WMMA issue ratio.
__global__ __launch_bounds__(256) void gemm_bf16_kernel(const bf16_t* __restrict__ A,
                                                        const bf16_t* __restrict__ BT,
                                                        float* __restrict__ Cout,
                                                        int Mrows, int Ncols, int K,
                                                        const float* __restrict__ bias,
                                                        const float* __restrict__ resid,
                                                        float* __restrict__ amax) {
  const int tid = threadIdx.x, wave = tid >> 5, lane = tid & 31;
  const int lr = lane & 15, hi = lane >> 4;
  const int m0 = blockIdx.y * 256 + (wave >> 1) * 64;
  const int n0 = blockIdx.x * 128 + (wave & 1) * 64;
  v8f acc[4][4];
#pragma unroll
  for (int i = 0; i < 4; ++i)
#pragma unroll
    for (int j = 0; j < 4; ++j)
      acc[i][j] = v8f{0.f, 0.f, 0.f, 0.f, 0.f, 0.f, 0.f, 0.f};

  const bf16_t* ap[4];
  const bf16_t* bp[4];
#pragma unroll
  for (int i = 0; i < 4; ++i) {
    int ra = m0 + i * 16 + lr;
    if (ra > Mrows - 1) ra = Mrows - 1;
    ap[i] = A + (size_t)ra * K;
    bp[i] = BT + (size_t)(n0 + i * 16 + lr) * K;
  }
  const int kofs = hi * 8;

  for (int k0 = 0; k0 < K; k0 += 32) {
    v16bf af[4], bf[4];
#pragma unroll
    for (int i = 0; i < 4; ++i) af[i] = ldfrag(ap[i] + k0 + kofs);
#pragma unroll
    for (int j = 0; j < 4; ++j) bf[j] = ldfrag(bp[j] + k0 + kofs);
#pragma unroll
    for (int i = 0; i < 4; ++i)
#pragma unroll
      for (int j = 0; j < 4; ++j)
        acc[i][j] = wmma_bf(af[i], bf[j], acc[i][j]);
  }

  float lmax = 0.f;
#pragma unroll
  for (int i = 0; i < 4; ++i)
#pragma unroll
    for (int j = 0; j < 4; ++j)
#pragma unroll
      for (int r = 0; r < 8; ++r) {
        int m = m0 + i * 16 + r + hi * 8;
        int n = n0 + j * 16 + lr;
        if (m < Mrows) {
          float v = acc[i][j][r];
          if (bias)  v += bias[n];
          if (resid) v += resid[(size_t)m * Ncols + n];
          Cout[(size_t)m * Ncols + n] = v;
          lmax = fmaxf(lmax, fabsf(v));
        }
      }
  if (amax) atomicMax((int*)amax, __float_as_int(lmax));
}

// ---------- GEGLU GEMM: out[bf16, MxFF] = (A*W1a+b1a) * gelu(A*W1g+b1g) ----------
__global__ __launch_bounds__(256) void gemm_geglu_kernel(const bf16_t* __restrict__ A,
                                                         const bf16_t* __restrict__ BT, // [FF2][K]
                                                         const float* __restrict__ bias, // [FF2]
                                                         bf16_t* __restrict__ out,       // [M][FF]
                                                         int Mrows, int K, int NcolsHalf) {
  const int tid = threadIdx.x, wave = tid >> 5, lane = tid & 31;
  const int lr = lane & 15, hi = lane >> 4;
  const int m0 = blockIdx.y * 128 + (wave >> 1) * 32;
  const int n0 = blockIdx.x * 64 + (wave & 1) * 32;
  v8f acca[2][2], accg[2][2];
#pragma unroll
  for (int i = 0; i < 2; ++i)
#pragma unroll
    for (int j = 0; j < 2; ++j) {
      acca[i][j] = v8f{0.f, 0.f, 0.f, 0.f, 0.f, 0.f, 0.f, 0.f};
      accg[i][j] = v8f{0.f, 0.f, 0.f, 0.f, 0.f, 0.f, 0.f, 0.f};
    }
  const bf16_t* a0p = A + (size_t)(m0 + lr) * K;
  const bf16_t* a1p = A + (size_t)(m0 + 16 + lr) * K;
  const bf16_t* ba0 = BT + (size_t)(n0 + lr) * K;
  const bf16_t* ba1 = BT + (size_t)(n0 + 16 + lr) * K;
  const bf16_t* bg0 = BT + (size_t)(NcolsHalf + n0 + lr) * K;
  const bf16_t* bg1 = BT + (size_t)(NcolsHalf + n0 + 16 + lr) * K;
  const int kofs = hi * 8;

  for (int k0 = 0; k0 < K; k0 += 32) {
    v16bf a0 = ldfrag(a0p + k0 + kofs);
    v16bf a1 = ldfrag(a1p + k0 + kofs);
    v16bf fa0 = ldfrag(ba0 + k0 + kofs);
    v16bf fa1 = ldfrag(ba1 + k0 + kofs);
    v16bf fg0 = ldfrag(bg0 + k0 + kofs);
    v16bf fg1 = ldfrag(bg1 + k0 + kofs);
    acca[0][0] = wmma_bf(a0, fa0, acca[0][0]);
    acca[0][1] = wmma_bf(a0, fa1, acca[0][1]);
    acca[1][0] = wmma_bf(a1, fa0, acca[1][0]);
    acca[1][1] = wmma_bf(a1, fa1, acca[1][1]);
    accg[0][0] = wmma_bf(a0, fg0, accg[0][0]);
    accg[0][1] = wmma_bf(a0, fg1, accg[0][1]);
    accg[1][0] = wmma_bf(a1, fg0, accg[1][0]);
    accg[1][1] = wmma_bf(a1, fg1, accg[1][1]);
  }

#pragma unroll
  for (int i = 0; i < 2; ++i)
#pragma unroll
    for (int j = 0; j < 2; ++j)
#pragma unroll
      for (int r = 0; r < 8; ++r) {
        int m = m0 + i * 16 + r + hi * 8;
        int n = n0 + j * 16 + lr;
        float av = acca[i][j][r] + bias[n];
        float gv = accg[i][j][r] + bias[NcolsHalf + n];
        out[(size_t)m * NcolsHalf + n] = (bf16_t)(av * gelu_tanh(gv));
      }
}

// ---------- attention pass 1: per-row softmax stats (m, Z) + global max(1/Z) ----------
__global__ __launch_bounds__(256) void attn_pass1_kernel(const bf16_t* __restrict__ q,
                                                         const bf16_t* __restrict__ k,
                                                         float* __restrict__ statm,
                                                         float* __restrict__ statz,
                                                         float* __restrict__ gmax,
                                                         int nkeys, int kTiles, int kBatchStride) {
  __shared__ float sm[8][16][33];
  const int tid = threadIdx.x, wave = tid >> 5, lane = tid & 31;
  const int lr = lane & 15, hi = lane >> 4;
  const int b = blockIdx.z, h = blockIdx.y;
  const int q0 = blockIdx.x * 128 + wave * 16;
  const int colbase = h * DH_;

  float rm = -1e30f, rz = 0.f;
  const bf16_t* qrow = q + (size_t)(b * N_ + q0 + lr) * C_;

  for (int kt = 0; kt < kTiles; ++kt) {
    int kbase = kt * 32;
    v8f s0 = v8f{0.f, 0.f, 0.f, 0.f, 0.f, 0.f, 0.f, 0.f};
    v8f s1 = s0;
#pragma unroll
    for (int d0 = 0; d0 < DH_; d0 += 32) {
      int kb = colbase + d0 + hi * 8;
      v16bf af = ldfrag(qrow + kb);
      v16bf b0 = ldfrag(k + (size_t)(b * kBatchStride + kbase + lr) * C_ + kb);
      v16bf b1 = ldfrag(k + (size_t)(b * kBatchStride + kbase + 16 + lr) * C_ + kb);
      s0 = wmma_bf(af, b0, s0);
      s1 = wmma_bf(af, b1, s1);
    }
    int k0i = kbase + lr, k1i = kbase + 16 + lr;
#pragma unroll
    for (int r = 0; r < 8; ++r) {
      int qi = r + hi * 8;
      sm[wave][qi][lr]      = (k0i < nkeys) ? s0[r] * 0.125f : -1e30f;
      sm[wave][qi][16 + lr] = (k1i < nkeys) ? s1[r] * 0.125f : -1e30f;
    }
    __syncthreads();
    if (lane < 16) {
      float tmax = -1e30f;
      for (int j = 0; j < 32; ++j) tmax = fmaxf(tmax, sm[wave][lane][j]);
      float nm = fmaxf(rm, tmax);
      float zz = rz * __expf(rm - nm);
      for (int j = 0; j < 32; ++j) zz += __expf(sm[wave][lane][j] - nm);
      rm = nm; rz = zz;
    }
    __syncthreads();
  }
  if (lane < 16) {
    size_t sidx = ((size_t)(b * H_ + h)) * N_ + q0 + lane;
    statm[sidx] = rm;
    statz[sidx] = rz;
    atomicMax((int*)gmax, __float_as_int(1.0f / rz));
  }
}

// ---------- attention pass 2: probs (quantized, fq_zero) @ V -> bf16 out ----------
__global__ __launch_bounds__(256) void attn_pass2_kernel(const bf16_t* __restrict__ q,
                                                         const bf16_t* __restrict__ k,
                                                         const bf16_t* __restrict__ vT,
                                                         const float* __restrict__ statm,
                                                         const float* __restrict__ statz,
                                                         const float* __restrict__ gmax,
                                                         bf16_t* __restrict__ out,
                                                         int nkeys, int kTiles,
                                                         int kBatchStride, int vtld) {
  __shared__ bf16_t wt[8][16][32];
  __shared__ float smm[8][16], smz[8][16];
  const int tid = threadIdx.x, wave = tid >> 5, lane = tid & 31;
  const int lr = lane & 15, hi = lane >> 4;
  const int b = blockIdx.z, h = blockIdx.y;
  const int q0 = blockIdx.x * 128 + wave * 16;
  const int colbase = h * DH_;

  if (lane < 16) {
    size_t sidx = ((size_t)(b * H_ + h)) * N_ + q0 + lane;
    smm[wave][lane] = statm[sidx];
    smz[wave][lane] = statz[sidx];
  }
  __syncthreads();

  const float delta = fmaxf(gmax[0], 1e-8f) / 255.0f;
  const bf16_t* qrow = q + (size_t)(b * N_ + q0 + lr) * C_;

  v8f o[4];
#pragma unroll
  for (int dt = 0; dt < 4; ++dt) o[dt] = v8f{0.f, 0.f, 0.f, 0.f, 0.f, 0.f, 0.f, 0.f};

  for (int kt = 0; kt < kTiles; ++kt) {
    int kbase = kt * 32;
    v8f s0 = v8f{0.f, 0.f, 0.f, 0.f, 0.f, 0.f, 0.f, 0.f};
    v8f s1 = s0;
#pragma unroll
    for (int d0 = 0; d0 < DH_; d0 += 32) {
      int kb = colbase + d0 + hi * 8;
      v16bf af = ldfrag(qrow + kb);
      v16bf b0 = ldfrag(k + (size_t)(b * kBatchStride + kbase + lr) * C_ + kb);
      v16bf b1 = ldfrag(k + (size_t)(b * kBatchStride + kbase + 16 + lr) * C_ + kb);
      s0 = wmma_bf(af, b0, s0);
      s1 = wmma_bf(af, b1, s1);
    }
    int k0i = kbase + lr, k1i = kbase + 16 + lr;
#pragma unroll
    for (int r = 0; r < 8; ++r) {
      int qi = r + hi * 8;
      float mrow = smm[wave][qi], zrow = smz[wave][qi];
      float w0 = (k0i < nkeys) ? __expf(s0[r] * 0.125f - mrow) / zrow : 0.f;
      float w1 = (k1i < nkeys) ? __expf(s1[r] * 0.125f - mrow) / zrow : 0.f;
      w0 = fminf(fmaxf(rintf(w0 / delta), 0.f), 255.f) * delta;
      w1 = fminf(fmaxf(rintf(w1 / delta), 0.f), 255.f) * delta;
      wt[wave][qi][lr]      = (bf16_t)w0;
      wt[wave][qi][16 + lr] = (bf16_t)w1;
    }
    __syncthreads();
    // probs as A-fragment (16 queries x 32 keys) from LDS
    v16bf aw = ldfrag(&wt[wave][lr][hi * 8]);
#pragma unroll
    for (int dt = 0; dt < 4; ++dt) {
      const bf16_t* vp = vT + ((size_t)b * C_ + colbase + dt * 16 + lr) * vtld + kbase + hi * 8;
      v16bf bf = ldfrag(vp);
      o[dt] = wmma_bf(aw, bf, o[dt]);
    }
    __syncthreads();
  }

#pragma unroll
  for (int dt = 0; dt < 4; ++dt)
#pragma unroll
    for (int r = 0; r < 8; ++r) {
      int qi = r + hi * 8;
      out[(size_t)(b * N_ + q0 + qi) * C_ + colbase + dt * 16 + lr] = (bf16_t)o[dt][r];
    }
}

// ---------- V transpose (self): vT[b][c][n] = v[b*N+n][c] ----------
__global__ __launch_bounds__(256) void transpose_v_kernel(const bf16_t* __restrict__ v,
                                                          bf16_t* __restrict__ vT) {
  size_t idx = (size_t)blockIdx.x * 256 + threadIdx.x;
  size_t total = (size_t)B_ * C_ * N_;
  if (idx >= total) return;
  int n = (int)(idx % N_);
  int c = (int)((idx / N_) % C_);
  int b = (int)(idx / ((size_t)N_ * C_));
  vT[idx] = v[((size_t)b * N_ + n) * C_ + c];
}

// ---------- V transpose (cross, zero-padded keys): vT[b][c][mp] ----------
__global__ __launch_bounds__(256) void transpose_v2_kernel(const bf16_t* __restrict__ v,
                                                           bf16_t* __restrict__ vT) {
  size_t idx = (size_t)blockIdx.x * 256 + threadIdx.x;
  size_t total = (size_t)B_ * C_ * MP_;
  if (idx >= total) return;
  int mp = (int)(idx % MP_);
  int c = (int)((idx / MP_) % C_);
  int b = (int)(idx / ((size_t)MP_ * C_));
  vT[idx] = (mp < M_) ? v[((size_t)b * M_ + mp) * C_ + c] : (bf16_t)0.0f;
}

// ============================ host launch ============================
extern "C" void kernel_launch(void* const* d_in, const int* in_sizes, int n_in,
                              void* d_out, int out_size, void* d_ws, size_t ws_size,
                              hipStream_t stream) {
  (void)in_sizes; (void)n_in; (void)out_size; (void)ws_size;
  const float* x    = (const float*)d_in[0];
  const float* ctx  = (const float*)d_in[1];
  const float* ln1g = (const float*)d_in[2];
  const float* ln1b = (const float*)d_in[3];
  const float* ln2g = (const float*)d_in[4];
  const float* ln2b = (const float*)d_in[5];
  const float* ln3g = (const float*)d_in[6];
  const float* ln3b = (const float*)d_in[7];
  const float* Wq1  = (const float*)d_in[8];
  const float* Wk1  = (const float*)d_in[9];
  const float* Wv1  = (const float*)d_in[10];
  const float* Wo1  = (const float*)d_in[11];
  const float* bo1  = (const float*)d_in[12];
  const float* Wq2  = (const float*)d_in[13];
  const float* Wk2  = (const float*)d_in[14];
  const float* Wv2  = (const float*)d_in[15];
  const float* Wo2  = (const float*)d_in[16];
  const float* bo2  = (const float*)d_in[17];
  const float* Wff1 = (const float*)d_in[18];
  const float* bff1 = (const float*)d_in[19];
  const float* Wff2 = (const float*)d_in[20];
  const float* bff2 = (const float*)d_in[21];
  float* outp = (float*)d_out;

  char* wsc = (char*)d_ws;
  size_t off = 0;
  auto alloc = [&](size_t bytes) -> char* {
    char* p = wsc + off;
    off = (off + bytes + 255) & ~(size_t)255;
    return p;
  };
  bf16_t* wq1T  = (bf16_t*)alloc((size_t)C_ * C_ * 2);
  bf16_t* wk1T  = (bf16_t*)alloc((size_t)C_ * C_ * 2);
  bf16_t* wv1T  = (bf16_t*)alloc((size_t)C_ * C_ * 2);
  bf16_t* wo1T  = (bf16_t*)alloc((size_t)C_ * C_ * 2);
  bf16_t* wq2T  = (bf16_t*)alloc((size_t)C_ * C_ * 2);
  bf16_t* wk2T  = (bf16_t*)alloc((size_t)C_ * CC_ * 2);
  bf16_t* wv2T  = (bf16_t*)alloc((size_t)C_ * CC_ * 2);
  bf16_t* wo2T  = (bf16_t*)alloc((size_t)C_ * C_ * 2);
  bf16_t* wff1T = (bf16_t*)alloc((size_t)FF2_ * C_ * 2);
  bf16_t* wff2T = (bf16_t*)alloc((size_t)C_ * FF_ * 2);
  bf16_t* xa    = (bf16_t*)alloc((size_t)R_ * C_ * 2);
  bf16_t* qb    = (bf16_t*)alloc((size_t)R_ * C_ * 2);
  bf16_t* kb    = (bf16_t*)alloc((size_t)R_ * C_ * 2);
  bf16_t* vb    = (bf16_t*)alloc((size_t)R_ * C_ * 2);
  bf16_t* vT    = (bf16_t*)alloc((size_t)B_ * C_ * N_ * 2);
  bf16_t* ctxb  = (bf16_t*)alloc((size_t)RC_ * CC_ * 2);
  bf16_t* k2b   = (bf16_t*)alloc((size_t)640 * C_ * 2);
  bf16_t* v2b   = (bf16_t*)alloc((size_t)640 * C_ * 2);
  bf16_t* v2T   = (bf16_t*)alloc((size_t)B_ * C_ * MP_ * 2);
  bf16_t* aob   = (bf16_t*)alloc((size_t)R_ * C_ * 2);
  bf16_t* ggb   = (bf16_t*)alloc((size_t)R_ * FF_ * 2);
  float*  f32s  = (float*)alloc((size_t)R_ * C_ * 4);
  float*  xcur  = (float*)alloc((size_t)R_ * C_ * 4);
  float*  statm = (float*)alloc((size_t)B_ * H_ * N_ * 4);
  float*  statz = (float*)alloc((size_t)B_ * H_ * N_ * 4);
  float*  scal  = (float*)alloc(16 * 4);

  auto blks = [](size_t n) { return (unsigned)((n + 255) / 256); };

  init_scal_kernel<<<1, 32, 0, stream>>>(scal);

  // weights -> transposed bf16
  wcvt_kernel<<<blks((size_t)C_ * C_), 256, 0, stream>>>(Wq1, wq1T, C_, C_);
  wcvt_kernel<<<blks((size_t)C_ * C_), 256, 0, stream>>>(Wk1, wk1T, C_, C_);
  wcvt_kernel<<<blks((size_t)C_ * C_), 256, 0, stream>>>(Wv1, wv1T, C_, C_);
  wcvt_kernel<<<blks((size_t)C_ * C_), 256, 0, stream>>>(Wo1, wo1T, C_, C_);
  wcvt_kernel<<<blks((size_t)C_ * C_), 256, 0, stream>>>(Wq2, wq2T, C_, C_);
  wcvt_kernel<<<blks((size_t)CC_ * C_), 256, 0, stream>>>(Wk2, wk2T, CC_, C_);
  wcvt_kernel<<<blks((size_t)CC_ * C_), 256, 0, stream>>>(Wv2, wv2T, CC_, C_);
  wcvt_kernel<<<blks((size_t)C_ * C_), 256, 0, stream>>>(Wo2, wo2T, C_, C_);
  wcvt_kernel<<<blks((size_t)C_ * FF2_), 256, 0, stream>>>(Wff1, wff1T, C_, FF2_);
  wcvt_kernel<<<blks((size_t)FF_ * C_), 256, 0, stream>>>(Wff2, wff2T, FF_, C_);

  const dim3 gC(C_ / 128, R_ / 256);         // 10 x 32
  const dim3 gAttn(N_ / 128, H_, B_);        // 8 x 20 x 8
  const size_t RC_elems = (size_t)R_ * C_;

  // ===== self attention =====
  ln_kernel<<<R_, 256, 0, stream>>>(x, ln1g, ln1b, xa, C_);
  gemm_bf16_kernel<<<gC, 256, 0, stream>>>(xa, wq1T, f32s, R_, C_, C_, nullptr, nullptr, scal + 0);
  quant_cvt_kernel<<<blks(RC_elems), 256, 0, stream>>>(f32s, qb, scal + 0, RC_elems);
  gemm_bf16_kernel<<<gC, 256, 0, stream>>>(xa, wk1T, f32s, R_, C_, C_, nullptr, nullptr, scal + 1);
  quant_cvt_kernel<<<blks(RC_elems), 256, 0, stream>>>(f32s, kb, scal + 1, RC_elems);
  gemm_bf16_kernel<<<gC, 256, 0, stream>>>(xa, wv1T, f32s, R_, C_, C_, nullptr, nullptr, scal + 2);
  quant_cvt_kernel<<<blks(RC_elems), 256, 0, stream>>>(f32s, vb, scal + 2, RC_elems);
  transpose_v_kernel<<<blks((size_t)B_ * C_ * N_), 256, 0, stream>>>(vb, vT);
  attn_pass1_kernel<<<gAttn, 256, 0, stream>>>(qb, kb, statm, statz, scal + 3, N_, N_ / 32, N_);
  attn_pass2_kernel<<<gAttn, 256, 0, stream>>>(qb, kb, vT, statm, statz, scal + 3, aob, N_, N_ / 32, N_, N_);
  gemm_bf16_kernel<<<gC, 256, 0, stream>>>(aob, wo1T, xcur, R_, C_, C_, bo1, x, nullptr);

  // ===== cross attention =====
  ln_kernel<<<R_, 256, 0, stream>>>(xcur, ln2g, ln2b, xa, C_);
  gemm_bf16_kernel<<<gC, 256, 0, stream>>>(xa, wq2T, f32s, R_, C_, C_, nullptr, nullptr, scal + 4);
  quant_cvt_kernel<<<blks(RC_elems), 256, 0, stream>>>(f32s, qb, scal + 4, RC_elems);
  cvt_kernel<<<blks((size_t)RC_ * CC_), 256, 0, stream>>>(ctx, ctxb, (size_t)RC_ * CC_);
  const dim3 gKV(C_ / 128, (RC_ + 255) / 256);   // 10 x 3
  gemm_bf16_kernel<<<gKV, 256, 0, stream>>>(ctxb, wk2T, f32s, RC_, C_, CC_, nullptr, nullptr, scal + 5);
  quant_cvt_kernel<<<blks((size_t)RC_ * C_), 256, 0, stream>>>(f32s, k2b, scal + 5, (size_t)RC_ * C_);
  gemm_bf16_kernel<<<gKV, 256, 0, stream>>>(ctxb, wv2T, f32s, RC_, C_, CC_, nullptr, nullptr, scal + 6);
  quant_cvt_kernel<<<blks((size_t)RC_ * C_), 256, 0, stream>>>(f32s, v2b, scal + 6, (size_t)RC_ * C_);
  transpose_v2_kernel<<<blks((size_t)B_ * C_ * MP_), 256, 0, stream>>>(v2b, v2T);
  attn_pass1_kernel<<<gAttn, 256, 0, stream>>>(qb, k2b, statm, statz, scal + 7, M_, MP_ / 32, M_);
  attn_pass2_kernel<<<gAttn, 256, 0, stream>>>(qb, k2b, v2T, statm, statz, scal + 7, aob, M_, MP_ / 32, M_, MP_);
  gemm_bf16_kernel<<<gC, 256, 0, stream>>>(aob, wo2T, xcur, R_, C_, C_, bo2, xcur, nullptr);

  // ===== GEGLU feed-forward =====
  ln_kernel<<<R_, 256, 0, stream>>>(xcur, ln3g, ln3b, xa, C_);
  const dim3 gFF(FF_ / 64, R_ / 128);       // 80 x 64
  gemm_geglu_kernel<<<gFF, 256, 0, stream>>>(xa, wff1T, bff1, ggb, R_, C_, FF_);
  gemm_bf16_kernel<<<gC, 256, 0, stream>>>(ggb, wff2T, outp, R_, C_, FF_, bff2, xcur, nullptr);
}
